// MultiHeadAttention_75101798138281
// MI455X (gfx1250) — matrix operand order repaired and checked by reading, not compiled, not run on hardware
//
#include <hip/hip_runtime.h>
#include <hip/hip_bf16.h>

// MI455X / gfx1250: wave32, WMMA f32_16x16x32_f16 + double-buffered Tensor Data Mover staging.
typedef __attribute__((ext_vector_type(16))) _Float16     v16h;
typedef __attribute__((ext_vector_type(8)))  float        v8f;
typedef __attribute__((ext_vector_type(4)))  unsigned int u32x4;
typedef __attribute__((ext_vector_type(4)))  int          i32x4;
typedef __attribute__((ext_vector_type(8)))  int          i32x8;

#define DIM    384
#define NHEADS 12
#define HDIM   32
#define SEQ    2048
#define BATCH  4
#define MROWS  (BATCH * SEQ)   // 8192
#define QKVN   (3 * DIM)       // 1152
#define ATTN_SCALE 0.17677669529663687f  // 32^-0.5

#define WMMA_F16(a, b, c) \
  __builtin_amdgcn_wmma_f32_16x16x32_f16(false, (a), false, (b), (short)0, (c), false, false)

#if __has_builtin(__builtin_amdgcn_tensor_load_to_lds)
#define HAVE_TDM 1
#else
#define HAVE_TDM 0
#endif

union FragH {
  v16h v;
  _Float16 e[16];
  uint4 q[2];
};

// ---- Fragment loaders (layouts per CDNA5 ISA 7.12.2, wave32) ----

// A fragment 16x32 (MxK) from row-major f16 source (16B-aligned rows): two b128 loads.
__device__ __forceinline__ v16h load_a_f16(const _Float16* __restrict__ A, int ld, int lane) {
  const int m = lane & 15, kh = (lane >> 4) & 1;
  const _Float16* r = A + m * ld + kh * 8;
  FragH f;
  f.q[0] = *(const uint4*)(r);        // K = kh*8 .. kh*8+7
  f.q[1] = *(const uint4*)(r + 16);   // K = 16+kh*8 .. 16+kh*8+7
  return f.v;
}

// B fragment 32x16 (KxN) from an N-major source (each n has >=32 contiguous K elems).
__device__ __forceinline__ v16h load_b_kcontig(const _Float16* __restrict__ B, int ld, int lane) {
  const int n = lane & 15, kb = (lane >> 4) << 3;
  const _Float16* r = B + n * ld + kb;
  FragH f;
  f.q[0] = *(const uint4*)(r);
  f.q[1] = *(const uint4*)(r + 16);
  return f.v;
}

#if HAVE_TDM
// 2D TENSOR_LOAD_TO_LDS: tensor [td1][td0] (dim0 contiguous, stride st0 elements),
// tile [tile1][tile0] starting at gaddr, packed into LDS at lds_off. 2-byte elements.
// Descriptor packing per CDNA5 ISA.
__device__ __forceinline__ void tdm_load_2d(unsigned lds_off, unsigned long long gaddr,
                                            unsigned td0, unsigned td1, unsigned st0,
                                            unsigned tile0, unsigned tile1) {
  u32x4 g0;
  g0[0] = 1u;                                                    // count=1, user descriptor
  g0[1] = lds_off;                                               // lds_addr
  g0[2] = (unsigned)(gaddr & 0xFFFFFFFFull);                     // global_addr[31:0]
  g0[3] = (unsigned)((gaddr >> 32) & 0x1FFFFFFull) | (2u << 30); // global_addr[56:32] | type=2
  i32x8 g1;
  g1[0] = (int)(1u << 16);                                       // data_size=1 (2B), wg_mask=0
  g1[1] = (int)((td0 & 0xFFFFu) << 16);                          // tensor_dim0[15:0] @ bits 63:48
  g1[2] = (int)((td0 >> 16) | ((td1 & 0xFFFFu) << 16));          // td0 hi | td1 lo
  g1[3] = (int)((td1 >> 16) | (tile0 << 16));                    // td1 hi | tile_dim0
  g1[4] = (int)(tile1 & 0xFFFFu);                                // tile_dim1 (tile_dim2=0)
  g1[5] = (int)st0;                                              // tensor_dim0_stride[31:0]
  g1[6] = 0;                                                     // stride hi | dim1_stride lo
  g1[7] = 0;
  i32x4 z4 = {0, 0, 0, 0};
#if defined(__clang_major__) && (__clang_major__ >= 23)
  i32x8 z8 = {0, 0, 0, 0, 0, 0, 0, 0};
  __builtin_amdgcn_tensor_load_to_lds(g0, g1, z4, z4, z8, 0);
#else
  __builtin_amdgcn_tensor_load_to_lds(g0, g1, z4, z4, 0);
#endif
}
#endif  // HAVE_TDM

// ---- Kernel 0: one-shot f16 conversion: x (row-major) + weights (N-major [N][K]) ----
__global__ void __launch_bounds__(256)
convert_inputs(const float* __restrict__ x, const float* __restrict__ Wqkv,
               const float* __restrict__ Wp, _Float16* __restrict__ x16,
               _Float16* __restrict__ Wq16, _Float16* __restrict__ Wp16) {
  const int i = blockIdx.x * 256 + threadIdx.x;
  if (i < MROWS * DIM) x16[i] = (_Float16)x[i];
  if (i < QKVN * DIM) {  // Wq16[n*384 + k] = Wqkv[k*1152 + n]
    const int n = i / DIM, k = i % DIM;
    Wq16[i] = (_Float16)Wqkv[(size_t)k * QKVN + n];
  }
  if (i < DIM * DIM) {   // Wp16[n*384 + k] = Wp[k*384 + n]
    const int n = i / DIM, k = i % DIM;
    Wp16[i] = (_Float16)Wp[(size_t)k * DIM + n];
  }
}

// ---- Kernel 1: qkv = x @ W_qkv; q,k -> [B,H,N,32]; v -> [B,H,32,N] (transposed) ----
__global__ void __launch_bounds__(256)
qkv_kernel(const _Float16* __restrict__ x16, const _Float16* __restrict__ Wq16,
           _Float16* __restrict__ qb, _Float16* __restrict__ kb, _Float16* __restrict__ vb) {
  const int lane = threadIdx.x & 31, wave = threadIdx.x >> 5;
  const int m0 = blockIdx.y * 128 + wave * 16;
  const int n0 = blockIdx.x * 64;          // 16x64 tile per wave
  v8f acc[4] = {{}, {}, {}, {}};
#pragma unroll 4
  for (int kk = 0; kk < DIM; kk += 32) {
    const v16h a = load_a_f16(x16 + (size_t)m0 * DIM + kk, DIM, lane);
#pragma unroll
    for (int t = 0; t < 4; ++t) {
      const v16h b = load_b_kcontig(Wq16 + (size_t)(n0 + t * 16) * DIM + kk, DIM, lane);
      acc[t] = WMMA_F16(a, b, acc[t]);
    }
  }
  const int grow = (lane >> 4) << 3, gcol = lane & 15;
#pragma unroll
  for (int t = 0; t < 4; ++t) {
    const int n = n0 + t * 16 + gcol;
    const int s = n / DIM, rem = n % DIM, h = rem / HDIM, d = rem % HDIM;
    _Float16* dst = (s == 0) ? qb : ((s == 1) ? kb : vb);
    const float sc = (s == 0) ? ATTN_SCALE : 1.0f;
#pragma unroll
    for (int r = 0; r < 8; ++r) {
      const int m = m0 + grow + r;
      const int bbat = m >> 11, nseq = m & (SEQ - 1);
      const size_t head = (size_t)(bbat * NHEADS + h);
      const size_t idx = (s == 2) ? (head * HDIM + d) * SEQ + nseq    // v transposed [hd][seq]
                                  : (head * SEQ + nseq) * HDIM + d;   // q,k [seq][hd]
      dst[idx] = (_Float16)(acc[t][r] * sc);
    }
  }
}

// ---- Kernel 2: flash attention; 64-key blocks, double-buffered TDM staging ----
__global__ void __launch_bounds__(256)
attn_kernel(const _Float16* __restrict__ q, const _Float16* __restrict__ k,
            const _Float16* __restrict__ v, _Float16* __restrict__ ao) {
  __shared__ __align__(16) _Float16 Kt[2][64][32];     // [key][hd]  (B-frag for scores)
  __shared__ __align__(16) _Float16 Vt[2][32][64];     // [hd][key]  (B-frag for PV)
  __shared__ __align__(16) _Float16 Pb[8][16][64];     // per-wave P: D-layout -> A-layout

  const int lane = threadIdx.x & 31, wave = threadIdx.x >> 5;
  int id = blockIdx.x;
  const int qblk = id & 15; id >>= 4;
  const int h = id % NHEADS, b = id / NHEADS;
  const size_t head_off = (size_t)(b * NHEADS + h) * SEQ * HDIM;
  const _Float16* qh = q + head_off;
  const _Float16* kh = k + head_off;
  const _Float16* vh = v + head_off;   // transposed: [hd=32][SEQ]

  const int q0 = qblk * 128 + wave * 16;
  const v16h qf = load_a_f16(qh + (size_t)q0 * HDIM, HDIM, lane);  // resident all pass

  float mrow[8], lrow[8];
  v8f o0 = {}, o1 = {};
#pragma unroll
  for (int r = 0; r < 8; ++r) { mrow[r] = -1e30f; lrow[r] = 0.0f; }

  const int grow = (lane >> 4) << 3;   // rows this lane holds in C/D frags
  const int gcol = lane & 15;

#if HAVE_TDM
  // Issue tile 0 into buffer 0 before the loop; keep one tile in flight thereafter.
  if (wave == 0) {
    const unsigned kt0 = (unsigned)(uintptr_t)&Kt[0][0][0];
    const unsigned vt0 = (unsigned)(uintptr_t)&Vt[0][0][0];
    tdm_load_2d(kt0, (unsigned long long)(uintptr_t)kh,
                /*td0=*/HDIM, /*td1=*/SEQ, /*st0=*/HDIM, /*tile0=*/HDIM, /*tile1=*/64);
    tdm_load_2d(vt0, (unsigned long long)(uintptr_t)vh,
                /*td0=*/SEQ, /*td1=*/HDIM, /*st0=*/SEQ, /*tile0=*/64, /*tile1=*/HDIM);
  }
#endif

  const int nIter = SEQ / 64;
  for (int it = 0; it < nIter; ++it) {
    const int cur = it & 1;
    __syncthreads();  // all consumers of buffer cur^1 (iteration it-1) and Pb are done
#if HAVE_TDM
    if (wave == 0) {
      if (it + 1 < nIter) {
        // Prefetch tile it+1 into the other buffer while tile it is consumed.
        const int nxt = cur ^ 1;
        const int kb2 = (it + 1) * 64;
        tdm_load_2d((unsigned)(uintptr_t)&Kt[nxt][0][0],
                    (unsigned long long)(uintptr_t)(kh + (size_t)kb2 * HDIM),
                    HDIM, SEQ, HDIM, HDIM, 64);
        tdm_load_2d((unsigned)(uintptr_t)&Vt[nxt][0][0],
                    (unsigned long long)(uintptr_t)(vh + kb2),
                    SEQ, HDIM, SEQ, 64, HDIM);
        __builtin_amdgcn_s_wait_tensorcnt(2);  // in-order: <=2 left => tile it landed
      } else {
        __builtin_amdgcn_s_wait_tensorcnt(0);
      }
    }
#else
    {
      const int kbase = it * 64;
      const int t8 = threadIdx.x * 8;
      const int key = t8 >> 5, dd = t8 & 31;
      *(uint4*)&Kt[cur][key][dd] = *(const uint4*)(kh + (size_t)(kbase + key) * HDIM + dd);
      const int vr = t8 >> 6, vc = t8 & 63;
      *(uint4*)&Vt[cur][vr][vc] = *(const uint4*)(vh + (size_t)vr * SEQ + kbase + vc);
    }
#endif
    __syncthreads();  // tile `it` visible to every wave

    // scores: 16q x 64k via four WMMAs (hd==32 == WMMA K-depth; scale folded into q)
    const v8f zero = {};
    v8f s0 = WMMA_F16(qf, load_b_kcontig(&Kt[cur][0][0], 32, lane), zero);
    v8f s1 = WMMA_F16(qf, load_b_kcontig(&Kt[cur][16][0], 32, lane), zero);
    v8f s2 = WMMA_F16(qf, load_b_kcontig(&Kt[cur][32][0], 32, lane), zero);
    v8f s3 = WMMA_F16(qf, load_b_kcontig(&Kt[cur][48][0], 32, lane), zero);

    // online softmax; row reductions are butterflies over the 16-lane row group
#pragma unroll
    for (int r = 0; r < 8; ++r) {
      float mx = fmaxf(fmaxf(s0[r], s1[r]), fmaxf(s2[r], s3[r]));
      mx = fmaxf(mx, __shfl_xor(mx, 1, 32));
      mx = fmaxf(mx, __shfl_xor(mx, 2, 32));
      mx = fmaxf(mx, __shfl_xor(mx, 4, 32));
      mx = fmaxf(mx, __shfl_xor(mx, 8, 32));
      const float mnew  = fmaxf(mrow[r], mx);
      const float alpha = __expf(mrow[r] - mnew);
      const float p0 = __expf(s0[r] - mnew);
      const float p1 = __expf(s1[r] - mnew);
      const float p2 = __expf(s2[r] - mnew);
      const float p3 = __expf(s3[r] - mnew);
      float ps = (p0 + p1) + (p2 + p3);
      ps += __shfl_xor(ps, 1, 32);
      ps += __shfl_xor(ps, 2, 32);
      ps += __shfl_xor(ps, 4, 32);
      ps += __shfl_xor(ps, 8, 32);
      lrow[r] = lrow[r] * alpha + ps;
      mrow[r] = mnew;
      o0[r] *= alpha;
      o1[r] *= alpha;
      Pb[wave][grow + r][gcol]      = (_Float16)p0;   // D-layout -> LDS row-major
      Pb[wave][grow + r][gcol + 16] = (_Float16)p1;
      Pb[wave][grow + r][gcol + 32] = (_Float16)p2;
      Pb[wave][grow + r][gcol + 48] = (_Float16)p3;
    }
    __syncthreads();  // Pb ready

    // PV: P(16x64 keys) @ V(64x32 hd) as 2 K-steps x 2 hd halves, f32 accumulate
    const v16h pa0 = load_a_f16(&Pb[wave][0][0], 64, lane);   // keys 0..31
    const v16h pa1 = load_a_f16(&Pb[wave][0][32], 64, lane);  // keys 32..63
    o0 = WMMA_F16(pa0, load_b_kcontig(&Vt[cur][0][0], 64, lane), o0);    // hd 0-15,  k 0-31
    o0 = WMMA_F16(pa1, load_b_kcontig(&Vt[cur][0][32], 64, lane), o0);   // hd 0-15,  k 32-63
    o1 = WMMA_F16(pa0, load_b_kcontig(&Vt[cur][16][0], 64, lane), o1);   // hd 16-31, k 0-31
    o1 = WMMA_F16(pa1, load_b_kcontig(&Vt[cur][16][32], 64, lane), o1);  // hd 16-31, k 32-63
  }

  // normalize and write [B, N, H*hd] (heads concatenated)
#pragma unroll
  for (int r = 0; r < 8; ++r) {
    const int row = q0 + grow + r;
    const float inv = 1.0f / lrow[r];
    const size_t base = ((size_t)b * SEQ + row) * DIM + h * HDIM;
    ao[base + gcol]      = (_Float16)(o0[r] * inv);
    ao[base + gcol + 16] = (_Float16)(o1[r] * inv);
  }
}

// ---- Kernel 3: out = attn_out @ W_proj + b_proj (f32 output) ----
__global__ void __launch_bounds__(256)
proj_kernel(const _Float16* __restrict__ ao, const _Float16* __restrict__ Wp16,
            const float* __restrict__ bp, float* __restrict__ out) {
  const int lane = threadIdx.x & 31, wave = threadIdx.x >> 5;
  const int m0 = blockIdx.y * 128 + wave * 16;
  const int n0 = blockIdx.x * 64;
  v8f acc[4] = {{}, {}, {}, {}};
#pragma unroll 4
  for (int kk = 0; kk < DIM; kk += 32) {
    const v16h a = load_a_f16(ao + (size_t)m0 * DIM + kk, DIM, lane);
#pragma unroll
    for (int t = 0; t < 4; ++t) {
      const v16h b = load_b_kcontig(Wp16 + (size_t)(n0 + t * 16) * DIM + kk, DIM, lane);
      acc[t] = WMMA_F16(a, b, acc[t]);
    }
  }
  const int grow = (lane >> 4) << 3, gcol = lane & 15;
#pragma unroll
  for (int t = 0; t < 4; ++t) {
    const int n = n0 + t * 16 + gcol;
    const float bias = bp[n];
#pragma unroll
    for (int r = 0; r < 8; ++r) {
      const int m = m0 + grow + r;
      out[(size_t)m * DIM + n] = acc[t][r] + bias;
    }
  }
}

extern "C" void kernel_launch(void* const* d_in, const int* in_sizes, int n_in,
                              void* d_out, int out_size, void* d_ws, size_t ws_size,
                              hipStream_t stream) {
  const float* x    = (const float*)d_in[0];
  const float* Wqkv = (const float*)d_in[1];
  const float* Wp   = (const float*)d_in[2];
  const float* bp   = (const float*)d_in[3];
  float* out = (float*)d_out;

  // ws: x16 + q,k [B,H,N,32] + v [B,H,32,N] + attn_out [B*N,384] + f16 weights (~32.6 MB)
  const size_t per = (size_t)BATCH * NHEADS * SEQ * HDIM;  // 3,145,728 elems (== MROWS*DIM)
  _Float16* ws   = (_Float16*)d_ws;
  _Float16* x16  = ws;
  _Float16* qb   = ws + per;
  _Float16* kb   = ws + 2 * per;
  _Float16* vb   = ws + 3 * per;
  _Float16* ao   = ws + 4 * per;
  _Float16* Wq16 = ws + 5 * per;
  _Float16* Wp16 = Wq16 + (size_t)QKVN * DIM;

  convert_inputs<<<dim3((MROWS * DIM + 255) / 256), 256, 0, stream>>>(x, Wqkv, Wp,
                                                                     x16, Wq16, Wp16);
  qkv_kernel<<<dim3(QKVN / 64, MROWS / 128), 256, 0, stream>>>(x16, Wq16, qb, kb, vb);
  attn_kernel<<<dim3(BATCH * NHEADS * (SEQ / 128)), 256, 0, stream>>>(qb, kb, vb, ao);
  proj_kernel<<<dim3(DIM / 64, MROWS / 128), 256, 0, stream>>>(ao, Wp16, bp, out);
}
